// Classifier_3934190043587
// MI455X (gfx1250) — matrix operand order repaired and checked by reading, not compiled, not run on hardware
//
#include <hip/hip_runtime.h>
#include <hip/hip_bf16.h>
#include <math.h>

typedef __attribute__((ext_vector_type(16))) _Float16 v16h;
typedef __attribute__((ext_vector_type(8)))  _Float16 v8h;
typedef __attribute__((ext_vector_type(8)))  float    v8f;

#define J_    10
#define B_    256
#define M_    200
#define CON_H 128
#define GATES 384   // 3 * CON_H

__device__ __forceinline__ float sigmoidf_(float x) { return 1.0f / (1.0f + __expf(-x)); }

// ---------------------------------------------------------------------------
// Kernel 1: h_jet = elu(x_jet @ W_jet + b_jet)   [J,B,4] @ [4,64] -> [J,B,64]
// ---------------------------------------------------------------------------
__global__ void jet_feat_kernel(const float* __restrict__ xj,
                                const float* __restrict__ Wj,
                                const float* __restrict__ bj,
                                float* __restrict__ hjet) {
    int idx = blockIdx.x * blockDim.x + threadIdx.x;
    if (idx >= J_ * B_ * 64) return;
    int row = idx >> 6, c = idx & 63;
    float acc = bj[c];
#pragma unroll
    for (int d = 0; d < 4; ++d) acc = fmaf(xj[row * 4 + d], Wj[d * 64 + c], acc);
    hjet[idx] = acc > 0.0f ? acc : (__expf(acc) - 1.0f);
}

// ---------------------------------------------------------------------------
// Kernel 2: constituent GRU (hidden=128) over 200 steps via WMMA f16->f32.
// 16 sequences per block; the wave's Whh slice lives in REGISTERS (12 v16h
// B-operand tiles, loaded once); h double-buffered in LDS as f16 (rows padded
// to 136 halves = 272B -> bank-conflict-free A fetches).
// Wave w owns column tile w of each gate region (r: tile w, z: w+8, n: w+16).
// ---------------------------------------------------------------------------
__global__ __launch_bounds__(256)
void con_gru_kernel(const float* __restrict__ x_kin,    // [J,B,M,3]
                    const int*   __restrict__ x_type,   // [J,B,M]
                    const int*   __restrict__ con_mask, // [J,B]
                    const float* __restrict__ emb,      // [9,3]
                    const float* __restrict__ Wih,      // [6,384]
                    const float* __restrict__ Whh,      // [128,384]
                    const float* __restrict__ bih,      // [384]
                    const float* __restrict__ bhh,      // [384]
                    float* __restrict__ hcon)           // [J*B,128]
{
    __shared__ __align__(16) _Float16 ldsH[2][16][136]; // double-buffered h (8.5 KB)
    __shared__ float ldsX[16][6];                       // per-step inputs

    const int tid  = threadIdx.x;
    const int lane = tid & 31;
    const int w    = tid >> 5;          // wave 0..7
    const int hi   = (lane >> 4) & 1;   // lane half
    const int nn   = lane & 15;
    const int seq_base = blockIdx.x * 16;

    // --- hidden column this lane owns, per gate region ---
    const int ch = (w << 4) + nn;            // 0..127
    const int cr = ch, cz = 128 + ch, cn = 256 + ch;

    // --- B-operand tiles resident in registers for the whole sequence ---
    // WMMA f16 B layout (32x16, K-major per lane): lane holds column n,
    // K = kc*32 + (p<8 ? hi*8 + p : 16 + hi*8 + p-8), p = 0..15 packed 2-per-VGPR.
    v16h Br[4], Bz[4], Bn[4];
    {
        const int kbase = hi * 16;
#pragma unroll
        for (int kc = 0; kc < 4; ++kc) {
#pragma unroll
            for (int p = 0; p < 16; ++p) {
                const float* col = Whh + (kc * 32 + (p < 8 ? (kbase >> 1) + p : 16 + (kbase >> 1) + p - 8)) * GATES;
                Br[kc][p] = (_Float16)col[cr];
                Bz[kc][p] = (_Float16)col[cz];
                Bn[kc][p] = (_Float16)col[cn];
            }
        }
    }

    // --- hoisted per-lane input-projection weights & biases ---
    float wr[6], wz[6], wn[6];
#pragma unroll
    for (int d = 0; d < 6; ++d) {
        wr[d] = Wih[d * GATES + cr];
        wz[d] = Wih[d * GATES + cz];
        wn[d] = Wih[d * GATES + cn];
    }
    const float bir = bih[cr], biz = bih[cz], bin_ = bih[cn];
    const float bhr = bhh[cr], bhz = bhh[cz], bhn  = bhh[cn];

    int maskv[8];
#pragma unroll
    for (int v = 0; v < 8; ++v) maskv[v] = con_mask[seq_base + v + hi * 8];

    // zero both h buffers
    for (int i = tid; i < 2 * 16 * 136; i += 256)
        ((_Float16*)ldsH)[i] = (_Float16)0.0f;
    __syncthreads();

    for (int t = 0; t < M_; ++t) {
        const int cur = t & 1, nxt = cur ^ 1;

        // --- stage x_t (3 kin + 3 embedded type dims) for the 16 sequences ---
        if (tid < 96) {
            int s = tid / 6, d = tid - s * 6;
            int gi = (seq_base + s) * M_ + t;
            float xv = (d < 3) ? x_kin[gi * 3 + d] : emb[x_type[gi] * 3 + (d - 3)];
            ldsX[s][d] = xv;
            // prefetch inputs a few steps ahead (gfx1250 global_prefetch_b8)
            if (t + 8 < M_) {
                int gi2 = gi + 8;
                if (d < 3) __builtin_prefetch(&x_kin[gi2 * 3 + d], 0, 1);
                else       __builtin_prefetch(&x_type[gi2], 0, 1);
            }
        }
        __syncthreads();   // ldsX ready; prev-step h writes visible

        // --- per-lane input projection (K=6, VALU) ---
        float xr[8], xz[8], xn[8];
#pragma unroll
        for (int v = 0; v < 8; ++v) {
            int m = v + hi * 8;
            float x0 = ldsX[m][0], x1 = ldsX[m][1], x2 = ldsX[m][2];
            float x3 = ldsX[m][3], x4 = ldsX[m][4], x5 = ldsX[m][5];
            xr[v] = bir  + x0*wr[0] + x1*wr[1] + x2*wr[2] + x3*wr[3] + x4*wr[4] + x5*wr[5];
            xz[v] = biz  + x0*wz[0] + x1*wz[1] + x2*wz[2] + x3*wz[3] + x4*wz[4] + x5*wz[5];
            xn[v] = bin_ + x0*wn[0] + x1*wn[1] + x2*wn[2] + x3*wn[3] + x4*wn[4] + x5*wn[5];
        }

        // --- recurrent GEMM: hp = h @ Whh, 12 register-resident WMMAs/step ---
        v8f ar  = {0, 0, 0, 0, 0, 0, 0, 0};
        v8f az  = {0, 0, 0, 0, 0, 0, 0, 0};
        v8f an_ = {0, 0, 0, 0, 0, 0, 0, 0};
        const _Float16* hrow = &ldsH[cur][lane & 15][0];
#pragma unroll
        for (int kc = 0; kc < 4; ++kc) {
            // A operand (16x32 f16) from current hidden state, conflict-free
            v8h lo  = *(const v8h*)(hrow + kc * 32 + hi * 8);
            v8h hi8 = *(const v8h*)(hrow + kc * 32 + 16 + hi * 8);
            v16h A = __builtin_shufflevector(lo, hi8,
                     0, 1, 2, 3, 4, 5, 6, 7, 8, 9, 10, 11, 12, 13, 14, 15);
            ar  = __builtin_amdgcn_wmma_f32_16x16x32_f16(false, A, false, Br[kc],
                                                         (short)0, ar,  false, false);
            az  = __builtin_amdgcn_wmma_f32_16x16x32_f16(false, A, false, Bz[kc],
                                                         (short)0, az,  false, false);
            an_ = __builtin_amdgcn_wmma_f32_16x16x32_f16(false, A, false, Bn[kc],
                                                         (short)0, an_, false, false);
        }

        // --- gate math + hidden update + masked gather to global ---
#pragma unroll
        for (int v = 0; v < 8; ++v) {
            int m = v + hi * 8;
            float r  = sigmoidf_(xr[v] + ar[v] + bhr);
            float z  = sigmoidf_(xz[v] + az[v] + bhz);
            float ng = tanhf(xn[v] + r * (an_[v] + bhn));
            float hold = (float)ldsH[cur][m][ch];
            float hnew = (1.0f - z) * ng + z * hold;
            ldsH[nxt][m][ch] = (_Float16)hnew;
            if (maskv[v] == t) hcon[(seq_base + m) * CON_H + ch] = hnew;
        }
        __syncthreads();
    }
}

// ---------------------------------------------------------------------------
// Kernel 3: final GRU over jets (T=10, in=192, hidden=32) + softmax head.
// One wave per event; 8 events per block; Wih_f resident in LDS.
// ---------------------------------------------------------------------------
__global__ __launch_bounds__(256)
void fin_gru_kernel(const float* __restrict__ hjet,  // [J*B,64]
                    const float* __restrict__ hcon,  // [J*B,128]
                    const int*   __restrict__ jet_mask, // [B]
                    const float* __restrict__ Wih,   // [192,96]
                    const float* __restrict__ Whh,   // [32,96]
                    const float* __restrict__ bih,   // [96]
                    const float* __restrict__ bhh,   // [96]
                    const float* __restrict__ Wout,  // [32,2]
                    const float* __restrict__ bout,  // [2]
                    float* __restrict__ out)         // [B,2]
{
    __shared__ float ldsW[192 * 96];
    __shared__ float ldsWh[32 * 96];
    __shared__ float ldsBi[96], ldsBh[96];
    __shared__ float ldsXv[8][192];

    const int tid = threadIdx.x, lane = tid & 31, wv = tid >> 5;
    for (int i = tid; i < 192 * 96; i += 256) ldsW[i]  = Wih[i];
    for (int i = tid; i < 32 * 96;  i += 256) ldsWh[i] = Whh[i];
    if (tid < 96) { ldsBi[tid] = bih[tid]; ldsBh[tid] = bhh[tid]; }
    __syncthreads();

    const int ev = blockIdx.x * 8 + wv;
    const int jm = jet_mask[ev];
    const int g0 = lane, g1 = lane + 32, g2 = lane + 64;
    float h = 0.0f, hfin = 0.0f;

    for (int t = 0; t < J_; ++t) {
        const float* hj = hjet + (t * B_ + ev) * 64;
        const float* hc = hcon + (t * B_ + ev) * 128;
#pragma unroll
        for (int i = 0; i < 6; ++i) {
            int d = lane * 6 + i;
            ldsXv[wv][d] = (d < 64) ? hj[d] : hc[d - 64];
        }
        __syncthreads();
        float a0 = ldsBi[g0], a1 = ldsBi[g1], a2 = ldsBi[g2];
        for (int d = 0; d < 192; ++d) {
            float xv = ldsXv[wv][d];
            a0 = fmaf(xv, ldsW[d * 96 + g0], a0);
            a1 = fmaf(xv, ldsW[d * 96 + g1], a1);
            a2 = fmaf(xv, ldsW[d * 96 + g2], a2);
        }
        float p0 = ldsBh[g0], p1 = ldsBh[g1], p2 = ldsBh[g2];
#pragma unroll 4
        for (int k = 0; k < 32; ++k) {
            float hk = __shfl(h, k, 32);
            p0 = fmaf(hk, ldsWh[k * 96 + g0], p0);
            p1 = fmaf(hk, ldsWh[k * 96 + g1], p1);
            p2 = fmaf(hk, ldsWh[k * 96 + g2], p2);
        }
        float r = sigmoidf_(a0 + p0);
        float z = sigmoidf_(a1 + p1);
        float n = tanhf(a2 + r * p2);
        h = (1.0f - z) * n + z * h;
        if (t == jm) hfin = h;
        __syncthreads();
    }

    float p0 = hfin * Wout[lane * 2 + 0];
    float p1 = hfin * Wout[lane * 2 + 1];
#pragma unroll
    for (int off = 16; off > 0; off >>= 1) {
        p0 += __shfl_xor(p0, off, 32);
        p1 += __shfl_xor(p1, off, 32);
    }
    if (lane == 0) {
        float l0 = p0 + bout[0], l1 = p1 + bout[1];
        float mx = fmaxf(l0, l1);
        float e0 = __expf(l0 - mx), e1 = __expf(l1 - mx);
        float s = e0 + e1;
        out[ev * 2 + 0] = e0 / s;
        out[ev * 2 + 1] = e1 / s;
    }
}

// ---------------------------------------------------------------------------
extern "C" void kernel_launch(void* const* d_in, const int* in_sizes, int n_in,
                              void* d_out, int out_size, void* d_ws, size_t ws_size,
                              hipStream_t stream) {
    (void)in_sizes; (void)n_in; (void)out_size; (void)ws_size;
    const float* x_jet     = (const float*)d_in[0];   // [J,B,4]
    const float* x_con_kin = (const float*)d_in[1];   // [J,B,M,3]
    const int*   x_con_typ = (const int*)  d_in[2];   // [J,B,M]
    const int*   jet_mask  = (const int*)  d_in[3];   // [B]
    const int*   con_mask  = (const int*)  d_in[4];   // [J,B]
    const float* W_jet     = (const float*)d_in[5];   // [4,64]
    const float* b_jet     = (const float*)d_in[6];   // [64]
    const float* emb       = (const float*)d_in[7];   // [9,3]
    const float* Wih_c     = (const float*)d_in[8];   // [6,384]
    const float* Whh_c     = (const float*)d_in[9];   // [128,384]
    const float* bih_c     = (const float*)d_in[10];  // [384]
    const float* bhh_c     = (const float*)d_in[11];  // [384]
    const float* Wih_f     = (const float*)d_in[12];  // [192,96]
    const float* Whh_f     = (const float*)d_in[13];  // [32,96]
    const float* bih_f     = (const float*)d_in[14];  // [96]
    const float* bhh_f     = (const float*)d_in[15];  // [96]
    const float* W_out     = (const float*)d_in[16];  // [32,2]
    const float* b_out     = (const float*)d_in[17];  // [2]
    float* out = (float*)d_out;

    float* hjet = (float*)d_ws;                 // [J*B, 64]
    float* hcon = hjet + (size_t)J_ * B_ * 64;  // [J*B, 128]

    jet_feat_kernel<<<(J_ * B_ * 64 + 255) / 256, 256, 0, stream>>>(
        x_jet, W_jet, b_jet, hjet);

    con_gru_kernel<<<(J_ * B_) / 16, 256, 0, stream>>>(
        x_con_kin, x_con_typ, con_mask, emb, Wih_c, Whh_c, bih_c, bhh_c, hcon);

    fin_gru_kernel<<<B_ / 8, 256, 0, stream>>>(
        hjet, hcon, jet_mask, Wih_f, Whh_f, bih_f, bhh_f, W_out, b_out, out);
}
